// GroupedQueryAttention_91104846283138
// MI455X (gfx1250) — matrix operand order repaired and checked by reading, not compiled, not run on hardware
//
#include <hip/hip_runtime.h>
#include <hip/hip_bf16.h>

typedef __attribute__((ext_vector_type(16))) _Float16 v16h;
typedef __attribute__((ext_vector_type(8)))  float    v8f;

#define B_   2
#define S_   2048
#define E_   2048
#define QH_  16
#define KVH_ 4
#define D_   128
#define NQKV 3072        // 2048 (Q) + 512 (K) + 512 (V)
#define ROWS (B_ * S_)   // 4096

#define WMMA_F16(a, b, c) \
  __builtin_amdgcn_wmma_f32_16x16x32_f16(false, (a), false, (b), (short)0, (c), false, false)

union H8 { uint4 u; _Float16 h[8]; };

// ---------------- Tensor Data Mover (CDNA5 TDM) ----------------
#if __has_builtin(__builtin_amdgcn_tensor_load_to_lds) && __has_builtin(__builtin_amdgcn_s_wait_tensorcnt)
#define USE_TDM 1
#else
#define USE_TDM 0
#endif

#if USE_TDM
typedef unsigned int u32x4 __attribute__((ext_vector_type(4)));
typedef int i32x8 __attribute__((ext_vector_type(8)));
typedef int i32x4 __attribute__((ext_vector_type(4)));

// 2D tile DMA: global (row-major, stride0 elements/row) -> LDS (packed rows).
// D# built per ISA 8.3/8.4: group0 = {count=1, lds_addr, global_addr, type=2},
// group1 = {data_size=2B, tensor_dim0/1, tile_dim0/1, tensor_dim0_stride}.
// This toolchain exposes the 6-arg builtin: (g0, g1, g2, g3, g4, cpol).
__device__ __forceinline__ void tdm_load_2d(const void* gptr, void* lptr,
                                            unsigned tile0, unsigned tile1,
                                            unsigned dim0, unsigned dim1,
                                            unsigned stride0) {
  unsigned long long ga = (unsigned long long)gptr;
  unsigned la = (unsigned)(unsigned long long)(__attribute__((address_space(3))) void*)lptr;
  u32x4 g0;
  g0[0] = 1u;                                   // count = 1 valid descriptor
  g0[1] = la;                                   // lds_addr
  g0[2] = (unsigned)ga;                         // global_addr[31:0]
  g0[3] = (unsigned)(ga >> 32) | (2u << 30);    // global_addr[56:32] | type=2
  i32x8 g1;
  g1[0] = (int)(1u << 16);                      // data_size = 1 (2 bytes)
  g1[1] = (int)(dim0 << 16);                    // tensor_dim0[15:0]
  g1[2] = (int)((dim0 >> 16) | (dim1 << 16));   // tensor_dim0[31:16] | tensor_dim1[15:0]
  g1[3] = (int)((dim1 >> 16) | (tile0 << 16));  // tensor_dim1[31:16] | tile_dim0
  g1[4] = (int)(tile1 & 0xffffu);               // tile_dim1 (tile_dim2 = 0)
  g1[5] = (int)stride0;                         // tensor_dim0_stride[31:0]
  g1[6] = 0;                                    // stride hi / dim1_stride lo
  g1[7] = 0;
  i32x4 z4 = {0, 0, 0, 0};
  i32x8 z8 = {0, 0, 0, 0, 0, 0, 0, 0};
  __builtin_amdgcn_tensor_load_to_lds(g0, g1, z4, z4, z8, 0);
}
#endif

// ---- WMMA fragment loaders (wave32, 16x16x32 f16; ISA 7.12.2 layouts) ----
__device__ __forceinline__ v16h load_a_frag(const _Float16* base, int ld, int m0, int k0) {
  const int lane = threadIdx.x & 31;
  const int m  = m0 + (lane & 15);
  const int kb = k0 + ((lane >> 4) << 3);
  const _Float16* p = base + m * ld + kb;
  union { v16h v; uint4 q[2]; } u;
  u.q[0] = *(const uint4*)(p);
  u.q[1] = *(const uint4*)(p + 16);
  return u.v;
}

// B operand read from an N-major ([n][k]) tile; K order matches load_a_frag (two 8-element runs at kb and kb+16).
__device__ __forceinline__ v16h load_bt_frag(const _Float16* base, int ld, int n0, int k0) {
  const int lane = threadIdx.x & 31;
  const int n  = n0 + (lane & 15);
  const int kb = k0 + ((lane >> 4) << 3);
  const _Float16* p = base + n * ld + kb;
  union { v16h v; uint4 q[2]; } u;
  u.q[0] = *(const uint4*)(p);
  u.q[1] = *(const uint4*)(p + 16);
  return u.v;
}

// ------------------------- elementwise preps -------------------------
__global__ void f32_to_f16_kernel(const float* __restrict__ in, _Float16* __restrict__ out, int n) {
  int i = blockIdx.x * blockDim.x + threadIdx.x;
  if (i < n) out[i] = (_Float16)in[i];
}

// [Wq|Wk|Wv] packed TRANSPOSED: out[n][k], n in [0,NQKV), k in [0,E)
__global__ void pack_wqkv_t_kernel(const float* __restrict__ Wq, const float* __restrict__ Wk,
                                   const float* __restrict__ Wv, _Float16* __restrict__ out, int n) {
  int i = blockIdx.x * blockDim.x + threadIdx.x;
  if (i >= n) return;
  int c = i / E_, k = i % E_;
  float v = (c < 2048) ? Wq[(size_t)k * 2048 + c]
          : (c < 2560) ? Wk[(size_t)k * 512 + (c - 2048)]
                       : Wv[(size_t)k * 512 + (c - 2560)];
  out[i] = (_Float16)v;
}

// dst[n][k] = src[k][n] (f32 -> f16)
__global__ void transpose_f32_to_f16_kernel(const float* __restrict__ src, _Float16* __restrict__ dst,
                                            int rows, int cols, int n) {
  int i = blockIdx.x * blockDim.x + threadIdx.x;
  if (i >= n) return;
  int c = i / rows, k = i % rows;
  dst[i] = (_Float16)src[(size_t)k * cols + c];
}

// ------------------------- 128x128 GEMM mainloop -------------------------
// C[ROWS][NQKV] = A[ROWS][E] @ Bt[NQKV][E]^T + bias, f16 out.
__global__ __launch_bounds__(256) void gemm_qkv_kernel(
    const _Float16* __restrict__ A, const _Float16* __restrict__ Bt,
    const float* __restrict__ bq, const float* __restrict__ bk, const float* __restrict__ bv,
    _Float16* __restrict__ Cout) {
  __shared__ _Float16 sA[128][64];
  __shared__ _Float16 sB[128][64];
  const int m0 = blockIdx.y * 128;
  const int n0 = blockIdx.x * 128;
  const int t = threadIdx.x;
  const int lane = t & 31;
  const int wave = t >> 5;
  const int mt = (wave & 3) * 32;
  const int nt = (wave >> 2) * 64;
  v8f acc[2][4];
  #pragma unroll
  for (int i = 0; i < 2; ++i)
    #pragma unroll
    for (int j = 0; j < 4; ++j) acc[i][j] = v8f{};

  for (int kk = 0; kk < E_; kk += 64) {
#if USE_TDM
    if (wave == 0) {
      tdm_load_2d(A + (size_t)m0 * E_ + kk, &sA[0][0], 64, 128, E_, 1u << 20, E_);
      __builtin_amdgcn_s_wait_tensorcnt(0);
    } else if (wave == 1) {
      tdm_load_2d(Bt + (size_t)n0 * E_ + kk, &sB[0][0], 64, 128, E_, 1u << 20, E_);
      __builtin_amdgcn_s_wait_tensorcnt(0);
    }
#else
    {
      int row = t >> 1, col = (t & 1) * 32;
      const uint4* sa = (const uint4*)(A + (size_t)(m0 + row) * E_ + kk + col);
      const uint4* sb = (const uint4*)(Bt + (size_t)(n0 + row) * E_ + kk + col);
      uint4* da = (uint4*)&sA[row][col];
      uint4* db = (uint4*)&sB[row][col];
      #pragma unroll
      for (int j = 0; j < 4; ++j) { da[j] = sa[j]; db[j] = sb[j]; }
    }
#endif
    __syncthreads();
    #pragma unroll
    for (int kc = 0; kc < 64; kc += 32) {
      v16h a0 = load_a_frag(&sA[0][0], 64, mt, kc);
      v16h a1 = load_a_frag(&sA[0][0], 64, mt + 16, kc);
      #pragma unroll
      for (int j = 0; j < 4; ++j) {
        v16h b = load_bt_frag(&sB[0][0], 64, nt + 16 * j, kc);
        acc[0][j] = WMMA_F16(a0, b, acc[0][j]);
        acc[1][j] = WMMA_F16(a1, b, acc[1][j]);
      }
    }
    __syncthreads();
  }
  #pragma unroll
  for (int i = 0; i < 2; ++i) {
    int mb = m0 + mt + 16 * i + ((lane >> 4) << 3);
    #pragma unroll
    for (int j = 0; j < 4; ++j) {
      int ncol = n0 + nt + 16 * j + (lane & 15);
      float bias = (ncol < 2048) ? bq[ncol] : (ncol < 2560 ? bk[ncol - 2048] : bv[ncol - 2560]);
      #pragma unroll
      for (int r = 0; r < 8; ++r)
        Cout[(size_t)(mb + r) * NQKV + ncol] = (_Float16)(acc[i][j][r] + bias);
    }
  }
}

// out[ROWS][E] = A[ROWS][E] @ Bt[E][E]^T + bo, f32 out.
__global__ __launch_bounds__(256) void gemm_out_kernel(
    const _Float16* __restrict__ A, const _Float16* __restrict__ Bt,
    const float* __restrict__ bo, float* __restrict__ Cout) {
  __shared__ _Float16 sA[128][64];
  __shared__ _Float16 sB[128][64];
  const int m0 = blockIdx.y * 128;
  const int n0 = blockIdx.x * 128;
  const int t = threadIdx.x;
  const int lane = t & 31;
  const int wave = t >> 5;
  const int mt = (wave & 3) * 32;
  const int nt = (wave >> 2) * 64;
  v8f acc[2][4];
  #pragma unroll
  for (int i = 0; i < 2; ++i)
    #pragma unroll
    for (int j = 0; j < 4; ++j) acc[i][j] = v8f{};

  for (int kk = 0; kk < E_; kk += 64) {
#if USE_TDM
    if (wave == 0) {
      tdm_load_2d(A + (size_t)m0 * E_ + kk, &sA[0][0], 64, 128, E_, 1u << 20, E_);
      __builtin_amdgcn_s_wait_tensorcnt(0);
    } else if (wave == 1) {
      tdm_load_2d(Bt + (size_t)n0 * E_ + kk, &sB[0][0], 64, 128, E_, 1u << 20, E_);
      __builtin_amdgcn_s_wait_tensorcnt(0);
    }
#else
    {
      int row = t >> 1, col = (t & 1) * 32;
      const uint4* sa = (const uint4*)(A + (size_t)(m0 + row) * E_ + kk + col);
      const uint4* sb = (const uint4*)(Bt + (size_t)(n0 + row) * E_ + kk + col);
      uint4* da = (uint4*)&sA[row][col];
      uint4* db = (uint4*)&sB[row][col];
      #pragma unroll
      for (int j = 0; j < 4; ++j) { da[j] = sa[j]; db[j] = sb[j]; }
    }
#endif
    __syncthreads();
    #pragma unroll
    for (int kc = 0; kc < 64; kc += 32) {
      v16h a0 = load_a_frag(&sA[0][0], 64, mt, kc);
      v16h a1 = load_a_frag(&sA[0][0], 64, mt + 16, kc);
      #pragma unroll
      for (int j = 0; j < 4; ++j) {
        v16h b = load_bt_frag(&sB[0][0], 64, nt + 16 * j, kc);
        acc[0][j] = WMMA_F16(a0, b, acc[0][j]);
        acc[1][j] = WMMA_F16(a1, b, acc[1][j]);
      }
    }
    __syncthreads();
  }
  #pragma unroll
  for (int i = 0; i < 2; ++i) {
    int mb = m0 + mt + 16 * i + ((lane >> 4) << 3);
    #pragma unroll
    for (int j = 0; j < 4; ++j) {
      int ncol = n0 + nt + 16 * j + (lane & 15);
      float bias = bo[ncol];
      #pragma unroll
      for (int r = 0; r < 8; ++r)
        Cout[(size_t)(mb + r) * E_ + ncol] = acc[i][j][r] + bias;
    }
  }
}

// ------------------------- RoPE + head-major scatter -------------------------
__global__ void rope_scatter_kernel(const _Float16* __restrict__ qkv,
                                    const float* __restrict__ sn, const float* __restrict__ cs,
                                    _Float16* __restrict__ Out, int nheads, int col_off, int total) {
  int i = blockIdx.x * blockDim.x + threadIdx.x;
  if (i >= total) return;
  int d = i & 63;
  int h = (i >> 6) % nheads;
  int s = (i / (64 * nheads)) % S_;
  int b = i / (64 * nheads * S_);
  const _Float16* row = qkv + (size_t)(b * S_ + s) * NQKV + col_off + h * D_;
  float x1 = (float)row[d], x2 = (float)row[d + 64];
  float sv = sn[(size_t)(b * S_ + s) * 64 + d];
  float cv = cs[(size_t)(b * S_ + s) * 64 + d];
  _Float16* orow = Out + (((size_t)b * nheads + h) * S_ + s) * D_;
  orow[d]      = (_Float16)(x1 * cv - x2 * sv);
  orow[d + 64] = (_Float16)(x2 * cv + x1 * sv);
}

// Vt[b][h][d][s] = V projection (transposed for the P*V B-operand)
__global__ void vt_scatter_kernel(const _Float16* __restrict__ qkv, _Float16* __restrict__ Vt, int total) {
  int i = blockIdx.x * blockDim.x + threadIdx.x;
  if (i >= total) return;
  int s = i % S_;
  int d = (i / S_) % D_;
  int h = (i / (S_ * D_)) % KVH_;
  int b = i / (S_ * D_ * KVH_);
  Vt[i] = qkv[(size_t)(b * S_ + s) * NQKV + 2560 + h * D_ + d];
}

// ------------------------- Flash attention -------------------------
__global__ __launch_bounds__(256) void attention_kernel(
    const _Float16* __restrict__ Q, const _Float16* __restrict__ K,
    const _Float16* __restrict__ Vt, const float* __restrict__ sinks,
    _Float16* __restrict__ AO) {
  __shared__ _Float16 sQ[64][128];
  __shared__ _Float16 sK[64][128];
  __shared__ _Float16 sVt[128][64];
  __shared__ _Float16 sP[64][64];
  __shared__ float sMax[64][2];
  __shared__ float sSum[64][2];

  const int q0 = blockIdx.x * 64;
  const int bh = blockIdx.y;
  const int b = bh / QH_, h = bh % QH_;
  const int kvh = h / (QH_ / KVH_);
  const int t = threadIdx.x;
  const int lane = t & 31;
  const int wave = t >> 5;
  const int mt = (wave & 3) * 16;
  const int g  = wave >> 2;
  const int rbase = mt + ((lane >> 4) << 3);

  const _Float16* Qbase = Q + (((size_t)b * QH_ + h) * S_ + q0) * D_;
  const _Float16* Kbase = K + (((size_t)b * KVH_ + kvh) * S_) * D_;
  const _Float16* Vtb   = Vt + (((size_t)b * KVH_ + kvh) * D_) * S_;

#if USE_TDM
  if (wave == 0) {
    tdm_load_2d(Qbase, &sQ[0][0], 128, 64, 128, 64, 128);
    __builtin_amdgcn_s_wait_tensorcnt(0);
  }
#else
  {
    int row = t >> 2, col = (t & 3) * 32;
    const uint4* src = (const uint4*)(Qbase + (size_t)row * D_ + col);
    uint4* dst = (uint4*)&sQ[row][col];
    #pragma unroll
    for (int j = 0; j < 4; ++j) dst[j] = src[j];
  }
#endif

  const float scale = 0.08838834764831845f;   // 1/sqrt(128)
  const float sink = sinks[h];                // softmax-invariant, kept for fidelity

  float Mrun[8], Lrun[8];
  #pragma unroll
  for (int r = 0; r < 8; ++r) { Mrun[r] = -3.0e38f; Lrun[r] = 0.f; }
  v8f o[4];
  #pragma unroll
  for (int i = 0; i < 4; ++i) o[i] = v8f{};

  for (int k0 = 0; k0 < S_; k0 += 64) {
#if USE_TDM
    if (wave == 0) {
      tdm_load_2d(Kbase + (size_t)k0 * D_, &sK[0][0], 128, 64, 128, 64, 128);
      __builtin_amdgcn_s_wait_tensorcnt(0);
    } else if (wave == 1) {
      tdm_load_2d(Vtb + k0, &sVt[0][0], 64, 128, S_ - k0, 128, S_);
      __builtin_amdgcn_s_wait_tensorcnt(0);
    }
#else
    {
      int row = t >> 2, col = (t & 3) * 32;
      const uint4* src = (const uint4*)(Kbase + (size_t)(k0 + row) * D_ + col);
      uint4* dst = (uint4*)&sK[row][col];
      #pragma unroll
      for (int j = 0; j < 4; ++j) dst[j] = src[j];
    }
    {
      int row = t >> 1, col = (t & 1) * 32;
      const uint4* src = (const uint4*)(Vtb + (size_t)row * S_ + k0 + col);
      uint4* dst = (uint4*)&sVt[row][col];
      #pragma unroll
      for (int j = 0; j < 4; ++j) dst[j] = src[j];
    }
#endif
    __syncthreads();

    // ---- S = Q K^T : two 16x16 tiles per wave over D=128
    v8f c0 = {}; v8f c1 = {};
    #pragma unroll
    for (int kc = 0; kc < 128; kc += 32) {
      v16h a  = load_a_frag(&sQ[0][0], 128, mt, kc);
      v16h b0 = load_bt_frag(&sK[0][0], 128, 32 * g, kc);
      v16h b1 = load_bt_frag(&sK[0][0], 128, 32 * g + 16, kc);
      c0 = WMMA_F16(a, b0, c0);
      c1 = WMMA_F16(a, b1, c1);
    }
    #pragma unroll
    for (int r = 0; r < 8; ++r) { c0[r] = c0[r] * scale + sink; c1[r] = c1[r] * scale + sink; }

    // ---- online softmax: row max partials (reduce across the 16-lane half)
    float pm[8];
    #pragma unroll
    for (int r = 0; r < 8; ++r) pm[r] = fmaxf(c0[r], c1[r]);
    #pragma unroll
    for (int mm = 8; mm >= 1; mm >>= 1)
      #pragma unroll
      for (int r = 0; r < 8; ++r) pm[r] = fmaxf(pm[r], __shfl_xor(pm[r], mm, 32));
    if ((lane & 15) == 0) {
      #pragma unroll
      for (int r = 0; r < 8; ++r) sMax[rbase + r][g] = pm[r];
    }
    __syncthreads();

    float fac[8], Mnew[8];
    #pragma unroll
    for (int r = 0; r < 8; ++r) {
      float mn = fmaxf(sMax[rbase + r][0], sMax[rbase + r][1]);
      Mnew[r] = fmaxf(Mrun[r], mn);
      fac[r] = __expf(Mrun[r] - Mnew[r]);
      Mrun[r] = Mnew[r];
    }

    // ---- P = exp(s - M) -> LDS (f16), partial row sums
    float ps[8];
    {
      int nc = 32 * g + (lane & 15);
      #pragma unroll
      for (int r = 0; r < 8; ++r) {
        float p0 = __expf(c0[r] - Mnew[r]);
        float p1 = __expf(c1[r] - Mnew[r]);
        sP[rbase + r][nc]      = (_Float16)p0;
        sP[rbase + r][nc + 16] = (_Float16)p1;
        ps[r] = p0 + p1;
      }
      #pragma unroll
      for (int mm = 8; mm >= 1; mm >>= 1)
        #pragma unroll
        for (int r = 0; r < 8; ++r) ps[r] += __shfl_xor(ps[r], mm, 32);
      if ((lane & 15) == 0) {
        #pragma unroll
        for (int r = 0; r < 8; ++r) sSum[rbase + r][g] = ps[r];
      }
    }
    __syncthreads();

    // ---- rescale running state, O += P V
    #pragma unroll
    for (int r = 0; r < 8; ++r)
      Lrun[r] = Lrun[r] * fac[r] + sSum[rbase + r][0] + sSum[rbase + r][1];
    #pragma unroll
    for (int dt = 0; dt < 4; ++dt)
      #pragma unroll
      for (int r = 0; r < 8; ++r) o[dt][r] *= fac[r];

    #pragma unroll
    for (int kc = 0; kc < 64; kc += 32) {
      v16h a = load_a_frag(&sP[0][0], 64, mt, kc);
      #pragma unroll
      for (int dt = 0; dt < 4; ++dt) {
        v16h bv = load_bt_frag(&sVt[0][0], 64, 64 * g + 16 * dt, kc);
        o[dt] = WMMA_F16(a, bv, o[dt]);
      }
    }
    __syncthreads();
  }

  float inv[8];
  #pragma unroll
  for (int r = 0; r < 8; ++r) inv[r] = (Lrun[r] > 0.f) ? 1.0f / Lrun[r] : 0.f;
  #pragma unroll
  for (int dt = 0; dt < 4; ++dt) {
    int dcol = 64 * g + 16 * dt + (lane & 15);
    #pragma unroll
    for (int r = 0; r < 8; ++r) {
      size_t row = (size_t)(b * S_ + q0 + rbase + r);
      AO[row * (QH_ * D_) + h * D_ + dcol] = (_Float16)(o[dt][r] * inv[r]);
    }
  }
}

// ------------------------- launch -------------------------
extern "C" void kernel_launch(void* const* d_in, const int* in_sizes, int n_in,
                              void* d_out, int out_size, void* d_ws, size_t ws_size,
                              hipStream_t stream) {
  (void)in_sizes; (void)n_in; (void)out_size; (void)ws_size;
  const float* x     = (const float*)d_in[0];
  const float* sn    = (const float*)d_in[1];
  const float* cs    = (const float*)d_in[2];
  const float* Wq    = (const float*)d_in[3];
  const float* bq    = (const float*)d_in[4];
  const float* Wk    = (const float*)d_in[5];
  const float* bk    = (const float*)d_in[6];
  const float* Wv    = (const float*)d_in[7];
  const float* bv    = (const float*)d_in[8];
  const float* Wo    = (const float*)d_in[9];
  const float* bo    = (const float*)d_in[10];
  const float* sinks = (const float*)d_in[11];
  float* out = (float*)d_out;

  char* ws = (char*)d_ws;
  size_t off = 0;
  auto alloc = [&](size_t bytes) -> void* {
    void* p = ws + off;
    off = (off + bytes + 255) & ~(size_t)255;
    return p;
  };
  _Float16* xh    = (_Float16*)alloc((size_t)ROWS * E_ * 2);
  _Float16* wqkvt = (_Float16*)alloc((size_t)NQKV * E_ * 2);
  _Float16* wot   = (_Float16*)alloc((size_t)E_ * E_ * 2);
  _Float16* qkv   = (_Float16*)alloc((size_t)ROWS * NQKV * 2);
  _Float16* Qb    = (_Float16*)alloc((size_t)B_ * QH_ * S_ * D_ * 2);
  _Float16* Kb    = (_Float16*)alloc((size_t)B_ * KVH_ * S_ * D_ * 2);
  _Float16* Vtb   = (_Float16*)alloc((size_t)B_ * KVH_ * S_ * D_ * 2);
  _Float16* AO    = (_Float16*)alloc((size_t)ROWS * E_ * 2);

  int n1 = ROWS * E_;
  f32_to_f16_kernel<<<(n1 + 255) / 256, 256, 0, stream>>>(x, xh, n1);
  int n2 = NQKV * E_;
  pack_wqkv_t_kernel<<<(n2 + 255) / 256, 256, 0, stream>>>(Wq, Wk, Wv, wqkvt, n2);
  int n3 = E_ * E_;
  transpose_f32_to_f16_kernel<<<(n3 + 255) / 256, 256, 0, stream>>>(Wo, wot, E_, E_, n3);

  dim3 g1(NQKV / 128, ROWS / 128);
  gemm_qkv_kernel<<<g1, 256, 0, stream>>>(xh, wqkvt, bq, bk, bv, qkv);

  int nq = B_ * S_ * QH_ * 64;
  rope_scatter_kernel<<<(nq + 255) / 256, 256, 0, stream>>>(qkv, sn, cs, Qb, QH_, 0, nq);
  int nk = B_ * S_ * KVH_ * 64;
  rope_scatter_kernel<<<(nk + 255) / 256, 256, 0, stream>>>(qkv, sn, cs, Kb, KVH_, QH_ * D_, nk);
  int nv = B_ * KVH_ * D_ * S_;
  vt_scatter_kernel<<<(nv + 255) / 256, 256, 0, stream>>>(qkv, Vtb, nv);

  dim3 g2(S_ / 64, B_ * QH_);
  attention_kernel<<<g2, 256, 0, stream>>>(Qb, Kb, Vtb, sinks, AO);

  dim3 g3(E_ / 128, ROWS / 128);
  gemm_out_kernel<<<g3, 256, 0, stream>>>(AO, wot, bo, out);
}
